// RavenFeatureRecurrentExitModel_35347580846292
// MI455X (gfx1250) — matrix operand order repaired and checked by reading, not compiled, not run on hardware
//
#include <hip/hip_runtime.h>
#include <math.h>

// ---------------------------------------------------------------------------
// Types for CDNA5 WMMA (wave32): v_wmma_f32_16x16x32_bf16
// ---------------------------------------------------------------------------
typedef __bf16 bf16;
typedef __attribute__((ext_vector_type(16))) __bf16 v16bf;
typedef __attribute__((ext_vector_type(8)))  float  v8f;
typedef __attribute__((ext_vector_type(4)))  unsigned u32x4;

// Problem constants
#define NEMBD 512
#define FEAT  32
#define DREC  544            // N_EMBD + FEAT
#define CAT   1088           // 2*DREC
#define HID   2176           // 2*CAT
#define HID2  4352           // 2*HID
#define NTOK  32768          // B*S = 8*4096
#define TSTEPS 8

// float -> bf16, round-to-nearest-even
__device__ __forceinline__ bf16 f2bf(float f) {
    unsigned u = __float_as_uint(f);
    unsigned r = (u + 0x7fffu + ((u >> 16) & 1u)) >> 16;
    unsigned short h = (unsigned short)r;
    bf16 o;
    __builtin_memcpy(&o, &h, 2);
    return o;
}

// Load a 32-byte WMMA fragment (16 bf16) from two 16B chunks at p and p+gap
__device__ __forceinline__ v16bf load_frag(const bf16* p, int gap_elems) {
    union { uint4 u[2]; v16bf v; } t;
    t.u[0] = *(const uint4*)(p);
    t.u[1] = *(const uint4*)(p + gap_elems);
    return t.v;
}

__device__ __forceinline__ v16bf pack_frag(u32x4 lo, u32x4 hi) {
    union { u32x4 u[2]; v16bf v; } t;
    t.u[0] = lo;
    t.u[1] = hi;
    return t.v;
}

__device__ __forceinline__ v8f wmma_bf16(v16bf a, v16bf b, v8f c) {
    return __builtin_amdgcn_wmma_f32_16x16x32_bf16(false, a, false, b,
                                                   (short)0, c, false, false);
}

// ---------------------------------------------------------------------------
// 0) zero recurrent state + output step-0 slice
// ---------------------------------------------------------------------------
__global__ void k_zero(float* __restrict__ rec, float* __restrict__ out) {
    size_t i      = (size_t)blockIdx.x * blockDim.x + threadIdx.x;
    size_t stride = (size_t)gridDim.x * blockDim.x;
    const size_t nrec = (size_t)NTOK * DREC;
    for (size_t idx = i; idx < nrec; idx += stride) rec[idx] = 0.0f;
    for (size_t m = i; m < NTOK; m += stride) {        // out[b,s,0,:] = 0
        out[m * (TSTEPS + 1) * 2 + 0] = 0.0f;
        out[m * (TSTEPS + 1) * 2 + 1] = 0.0f;
    }
}

// ---------------------------------------------------------------------------
// 1) weight transpose + bf16 convert: w[K][N] -> wt[N][K]
// ---------------------------------------------------------------------------
__global__ void k_transpose_bf16(const float* __restrict__ w, bf16* __restrict__ wt,
                                 int K, int N) {
    size_t idx = (size_t)blockIdx.x * blockDim.x + threadIdx.x;
    size_t total = (size_t)K * N;
    if (idx >= total) return;
    int k = (int)(idx % K);
    int n = (int)(idx / K);
    wt[idx] = f2bf(w[(size_t)k * N + n]);
}

// ---------------------------------------------------------------------------
// 2) fused concat + LayerNorm -> bf16 X  (one block of 256 threads per token)
// ---------------------------------------------------------------------------
__global__ void k_build_xln(const float* __restrict__ emb,
                            const float* __restrict__ feats,
                            const float* __restrict__ nw,
                            const float* __restrict__ nb,
                            const float* __restrict__ rec,
                            bf16* __restrict__ xln, int t) {
    const int m   = blockIdx.x;
    const int tid = threadIdx.x;
    __shared__ float red[256];

    const float* erow = emb  + (size_t)m * NEMBD;
    const float* frow = feats + ((size_t)m * TSTEPS + t) * FEAT;
    const float* rrow = rec  + (size_t)m * DREC;

    float vals[5];
    float sum = 0.0f;
#pragma unroll
    for (int j = 0; j < 5; ++j) {
        int i = tid + j * 256;
        float v = 0.0f;
        if (i < CAT) {
            if (i < NEMBD)      v = erow[i];
            else if (i < DREC)  v = frow[i - NEMBD];
            else                v = rrow[i - DREC];
        }
        vals[j] = v;
        sum += v;
    }
    red[tid] = sum;
    __syncthreads();
    for (int s = 128; s > 0; s >>= 1) { if (tid < s) red[tid] += red[tid + s]; __syncthreads(); }
    const float mu = red[0] * (1.0f / CAT);
    __syncthreads();

    float ss = 0.0f;
#pragma unroll
    for (int j = 0; j < 5; ++j) {
        int i = tid + j * 256;
        if (i < CAT) { float d = vals[j] - mu; ss += d * d; }
    }
    red[tid] = ss;
    __syncthreads();
    for (int s = 128; s > 0; s >>= 1) { if (tid < s) red[tid] += red[tid + s]; __syncthreads(); }
    const float rs = rsqrtf(red[0] * (1.0f / CAT) + 1e-5f);

    bf16* orow = xln + (size_t)m * CAT;
#pragma unroll
    for (int j = 0; j < 5; ++j) {
        int i = tid + j * 256;
        if (i < CAT) orow[i] = f2bf((vals[j] - mu) * rs * nw[i] + nb[i]);
    }
}

// ---------------------------------------------------------------------------
// 3) GEMM1 + SiLU gate:  gh = silu(Xln@Wa + ba) * (Xln@Wg + bg)
//    Workgroup = 8 waves, M-tile 64 rows staged in LDS via async-to-LDS,
//    double buffered (xor-toggled, branch-free, unroll-1 canonical body).
//    A fragments come back out of LDS via explicit ds_load_b128 asm (opaque
//    to the compiler, matching the asm write side). Wave w handles n-block
//    (nbg*8 + w): one 16-col a-tile and paired g-tile across 4 M-subtiles
//    (8 WMMA per K-step).
//    fcwt is [HID2][CAT] (transposed bf16), a cols [0,HID), g cols [HID,HID2).
// ---------------------------------------------------------------------------
__global__ void __launch_bounds__(256) k_gemm1_gate(const bf16* __restrict__ xln,
                                                    const bf16* __restrict__ fcwt,
                                                    const float* __restrict__ fcb,
                                                    bf16* __restrict__ gh) {
    __shared__ __align__(16) bf16 smem[2][64][32];   // 2 x 4KB A-tile buffers

    const int tid  = threadIdx.x;
    const int lane = tid & 31;
    const int wave = tid >> 5;
    const int mgroup = blockIdx.x / 17;              // 512 groups of 64 rows
    const int nbg    = blockIdx.x % 17;              // 17 groups of 8 n-blocks
    const int nb     = nbg * 8 + wave;               // 0..135
    const int mbase  = mgroup * 64;
    (void)smem;                                      // touched only via asm

    // per-thread async staging slice: one b128 (8 bf16) per thread per K-step
    const int srow = tid >> 2;                       // 0..63
    const int schk = (tid & 3) * 8;                  // half-element offset
    const bf16* gsrc = xln + (size_t)(mbase + srow) * CAT + schk;
    const unsigned ldsoff = (unsigned)(srow * 64 + (tid & 3) * 16); // bytes in buf

    const int kbA  = (lane < 16) ? 0 : 8;            // A lane K base
    const int kbB  = (lane < 16) ? 0 : 8;            // B lane K base
    const int ncol = nb * 16 + (lane & 15);
    const bf16* browa = fcwt + (size_t)ncol * CAT + kbB;
    const bf16* browg = fcwt + (size_t)(ncol + HID) * CAT + kbB;

    v8f acc_a[4] = {{}, {}, {}, {}};
    v8f acc_g[4] = {{}, {}, {}, {}};

    // prologue: issue K-step 0 into buffer 0
    {
        unsigned long long ga = (unsigned long long)gsrc;
        asm volatile("global_load_async_to_lds_b128 %0, %1, off"
                     :: "v"(ldsoff), "v"(ga) : "memory");
    }

    unsigned curbuf = 0;                             // current buffer byte offset
    const unsigned lrow = (unsigned)((lane & 15) * 64 + kbA * 2); // frag base in buf
    const int NK = CAT / 32;                         // 34 K-steps

#pragma unroll 1
    for (int ks = 0; ks < NK; ++ks) {
        asm volatile("s_wait_asynccnt 0" ::: "memory");
        __syncthreads();                             // current buffer ready for all

        // branch-free prefetch of next K-step into the other buffer.
        // (Final iteration prefetches <=48B past xln into the adjacent ws
        //  region -- valid memory, target buffer never read again.)
        {
            unsigned dst = ldsoff + (curbuf ^ 4096u);
            unsigned long long ga = (unsigned long long)(gsrc + 32);
            asm volatile("global_load_async_to_lds_b128 %0, %1, off"
                         :: "v"(dst), "v"(ga) : "memory");
        }

        // 4 A fragments (one per 16-row subtile) via explicit LDS loads
        u32x4 q0, q1, q2, q3, q4, q5, q6, q7;
        {
            unsigned fbase = curbuf + lrow;
            asm volatile(
                "ds_load_b128 %0, %8\n\t"
                "ds_load_b128 %1, %8 offset:32\n\t"
                "ds_load_b128 %2, %8 offset:1024\n\t"
                "ds_load_b128 %3, %8 offset:1056\n\t"
                "ds_load_b128 %4, %8 offset:2048\n\t"
                "ds_load_b128 %5, %8 offset:2080\n\t"
                "ds_load_b128 %6, %8 offset:3072\n\t"
                "ds_load_b128 %7, %8 offset:3104\n\t"
                "s_wait_dscnt 0"
                : "=v"(q0), "=v"(q1), "=v"(q2), "=v"(q3),
                  "=v"(q4), "=v"(q5), "=v"(q6), "=v"(q7)
                : "v"(fbase));
        }
        v16bf af0 = pack_frag(q0, q1);
        v16bf af1 = pack_frag(q2, q3);
        v16bf af2 = pack_frag(q4, q5);
        v16bf af3 = pack_frag(q6, q7);

        v16bf bfa = load_frag(browa, 16);            // K kbB..kbB+7, kbB+16..kbB+23
        v16bf bfg = load_frag(browg, 16);

        acc_a[0] = wmma_bf16(af0, bfa, acc_a[0]);
        acc_g[0] = wmma_bf16(af0, bfg, acc_g[0]);
        acc_a[1] = wmma_bf16(af1, bfa, acc_a[1]);
        acc_g[1] = wmma_bf16(af1, bfg, acc_g[1]);
        acc_a[2] = wmma_bf16(af2, bfa, acc_a[2]);
        acc_g[2] = wmma_bf16(af2, bfg, acc_g[2]);
        acc_a[3] = wmma_bf16(af3, bfa, acc_a[3]);
        acc_g[3] = wmma_bf16(af3, bfg, acc_g[3]);

        __syncthreads();                             // done reading current buffer

        curbuf ^= 4096u;
        gsrc  += 32;
        browa += 32;
        browg += 32;
    }

    // epilogue: bias + SiLU gate, write bf16 gated hidden
    const int n = lane & 15;
    const float ba = fcb[nb * 16 + n];
    const float bg = fcb[nb * 16 + n + HID];
#pragma unroll
    for (int s = 0; s < 4; ++s) {
        const int mrow0 = mbase + s * 16 + ((lane >> 4) << 3);
#pragma unroll
        for (int r = 0; r < 8; ++r) {
            float a  = acc_a[s][r] + ba;
            float g  = acc_g[s][r] + bg;
            float si = a * __builtin_amdgcn_rcpf(1.0f + __expf(-a));  // SiLU
            gh[(size_t)(mrow0 + r) * HID + nb * 16 + n] = f2bf(si * g);
        }
    }
}

// ---------------------------------------------------------------------------
// 4) GEMM2 + residual:  rec += gh @ proj_w + proj_b
//    pwt is [DREC][HID] (transposed bf16). Each wave: 64x16 tile
//    (4 M-subtiles sharing each B fragment), direct global fragments
//    (gh slice is L2-resident across the 34 n-blocks).
// ---------------------------------------------------------------------------
__global__ void __launch_bounds__(256) k_gemm2_update(const bf16* __restrict__ gh,
                                                      const bf16* __restrict__ pwt,
                                                      const float* __restrict__ pb,
                                                      float* __restrict__ rec) {
    const int NB   = DREC / 16;                      // 34 col-blocks
    const int wid  = (int)(((size_t)blockIdx.x * blockDim.x + threadIdx.x) >> 5);
    const int lane = threadIdx.x & 31;
    const int mgroup = wid / NB;                     // 512 groups of 64 rows
    const int nb     = wid % NB;
    const int mbase  = mgroup * 64;

    const int kbA  = (lane < 16) ? 0 : 8;
    const int kbB  = (lane < 16) ? 0 : 8;
    const int ncol = nb * 16 + (lane & 15);

    const bf16* arow0 = gh + (size_t)(mbase + (lane & 15)) * HID + kbA;
    const bf16* brow  = pwt + (size_t)ncol * HID + kbB;

    v8f acc[4] = {{}, {}, {}, {}};
#pragma unroll 1
    for (int k = 0; k < HID; k += 32) {
        v16bf bfr = load_frag(brow, 16);
#pragma unroll
        for (int s = 0; s < 4; ++s) {
            v16bf af = load_frag(arow0 + (size_t)s * 16 * HID + k, 16);
            acc[s] = wmma_bf16(af, bfr, acc[s]);
        }
        brow += 32;
    }

    const int n     = lane & 15;
    const float bias = pb[nb * 16 + n];
#pragma unroll
    for (int s = 0; s < 4; ++s) {
        const int mrow0 = mbase + s * 16 + ((lane >> 4) << 3);
#pragma unroll
        for (int r = 0; r < 8; ++r) {
            size_t idx = (size_t)(mrow0 + r) * DREC + nb * 16 + n;
            rec[idx] += acc[s][r] + bias;            // unique owner, no atomics
        }
    }
}

// ---------------------------------------------------------------------------
// 5) logits + 2-way softmax (one wave32 per token)
// ---------------------------------------------------------------------------
__global__ void k_logits(const float* __restrict__ rec,
                         const float* __restrict__ ow,
                         const float* __restrict__ ob,
                         float* __restrict__ out, int t) {
    const int lane = threadIdx.x & 31;
    const int m    = blockIdx.x * (blockDim.x >> 5) + (threadIdx.x >> 5);
    const float* r = rec + (size_t)m * DREC;

    float l0 = 0.0f, l1 = 0.0f;
    for (int i = lane; i < DREC; i += 32) {
        float v = r[i];
        l0 += v * ow[i * 2 + 0];
        l1 += v * ow[i * 2 + 1];
    }
#pragma unroll
    for (int off = 16; off > 0; off >>= 1) {
        l0 += __shfl_xor(l0, off, 32);
        l1 += __shfl_xor(l1, off, 32);
    }
    if (lane == 0) {
        l0 += ob[0]; l1 += ob[1];
        float mx = fmaxf(l0, l1);
        float e0 = __expf(l0 - mx), e1 = __expf(l1 - mx);
        float inv = __builtin_amdgcn_rcpf(e0 + e1);
        float* o = out + ((size_t)m * (TSTEPS + 1) + t + 1) * 2;
        o[0] = e0 * inv;
        o[1] = e1 * inv;
    }
}

// ---------------------------------------------------------------------------
// launch
// ---------------------------------------------------------------------------
extern "C" void kernel_launch(void* const* d_in, const int* in_sizes, int n_in,
                              void* d_out, int out_size, void* d_ws, size_t ws_size,
                              hipStream_t stream) {
    (void)in_sizes; (void)n_in; (void)out_size; (void)ws_size;
    const float* emb    = (const float*)d_in[0];
    const float* feats  = (const float*)d_in[1];
    const float* norm_w = (const float*)d_in[2];
    const float* norm_b = (const float*)d_in[3];
    const float* fc_w   = (const float*)d_in[4];
    const float* fc_b   = (const float*)d_in[5];
    const float* proj_w = (const float*)d_in[6];
    const float* proj_b = (const float*)d_in[7];
    const float* out_w  = (const float*)d_in[8];
    const float* out_b  = (const float*)d_in[9];
    float* out = (float*)d_out;

    // workspace layout (bytes, all 256B-aligned)
    char* ws = (char*)d_ws;
    float* rec  = (float*)(ws);                       // 32768*544*4   = 71,303,168
    bf16*  xln  = (bf16*)(ws + 71303168);             // 32768*1088*2  = 71,303,168
    bf16*  gh   = (bf16*)(ws + 142606336);            // 32768*2176*2  = 142,606,336
    bf16*  fcwt = (bf16*)(ws + 285212672);            // 4352*1088*2   =   9,469,952
    bf16*  pwt  = (bf16*)(ws + 294682624);            // 544*2176*2    =   2,367,488

    k_zero<<<4096, 256, 0, stream>>>(rec, out);
    k_transpose_bf16<<<(CAT * HID2 + 255) / 256, 256, 0, stream>>>(fc_w, fcwt, CAT, HID2);
    k_transpose_bf16<<<(HID * DREC + 255) / 256, 256, 0, stream>>>(proj_w, pwt, HID, DREC);

    const int g1_blocks = (NTOK / 64) * (HID / 16 / 8);     // 512*17 = 8704
    const int g2_blocks = (NTOK / 64) * (DREC / 16) / 8;    // 512*34/8 = 2176

    for (int t = 0; t < TSTEPS; ++t) {
        k_build_xln<<<NTOK, 256, 0, stream>>>(emb, feats, norm_w, norm_b, rec, xln, t);
        k_gemm1_gate<<<g1_blocks, 256, 0, stream>>>(xln, fcwt, fc_b, gh);
        k_gemm2_update<<<g2_blocks, 256, 0, stream>>>(gh, pwt, proj_b, rec);
        k_logits<<<4096, 256, 0, stream>>>(rec, out_w, out_b, out, t);
    }
}
